// SingleHeadAttention_18313740550789
// MI455X (gfx1250) — matrix operand order repaired and checked by reading, not compiled, not run on hardware
//
#include <hip/hip_runtime.h>

#define BB 4
#define SS 2048
#define DD 1024

typedef __bf16 bf16;
typedef __attribute__((ext_vector_type(16))) __bf16 v16bf;
typedef __attribute__((ext_vector_type(8)))  __bf16 v8bf;
typedef __attribute__((ext_vector_type(4)))  __bf16 v4bf;
typedef __attribute__((ext_vector_type(8)))  float  v8f;
typedef __attribute__((ext_vector_type(4)))  float  v4f;

static __device__ __forceinline__ v16bf cat8(v8bf lo, v8bf hi) {
  return __builtin_shufflevector(lo, hi, 0,1,2,3,4,5,6,7,8,9,10,11,12,13,14,15);
}

// ---------------- f32 -> bf16, 4 elements/thread ---------
__global__ __launch_bounds__(256) void cvt_bf16_kernel(const float* __restrict__ src,
                                                       bf16* __restrict__ dst, int n) {
  int i = (blockIdx.x * blockDim.x + threadIdx.x) * 4;
  if (i < n) {
    v4f f = *(const v4f*)(src + i);
    *(v4bf*)(dst + i) = __builtin_convertvector(f, v4bf);
  }
}

// ---------------- QKV projection GEMM --------------------
// Per-wave 32x64 tile: two 16x32 A fragments share each 32x16 B fragment
// (8 WMMAs per 12 b128 loads -> ~2x the FLOP/byte of a 16x64 tile).
// z = 0: Q = (x @ Wq^T) * 1/sqrt(D)       -> qb  [b*S + s][e]
// z = 1: K =  x @ Wk^T                    -> kb  [b*S + s][e]
// z = 2: V =  x @ Wv^T, stored transposed -> vt  [b][e][s]
__global__ __launch_bounds__(256) void qkv_kernel(const bf16* __restrict__ xbf,
                                                  const bf16* __restrict__ wq,
                                                  const bf16* __restrict__ wk,
                                                  const bf16* __restrict__ wvp,
                                                  bf16* __restrict__ qb,
                                                  bf16* __restrict__ kbuf,
                                                  bf16* __restrict__ vt) {
  const int lane = threadIdx.x & 31;
  const int wid  = threadIdx.x >> 5;
  const int m0   = blockIdx.x * 256 + wid * 32;   // row tile in B*S
  const int n0   = blockIdx.y * 64;               // col tile in D
  const int z    = blockIdx.z;
  const bf16* w  = (z == 0) ? wq : (z == 1) ? wk : wvp;

  const int am = lane & 15, ak = (lane >> 4) * 8;   // A: row per lane, split K
  const int bn = lane & 15, bk = (lane >> 4) * 8;   // B: col per lane, split K

  const bf16* aptr0 = xbf + (size_t)(m0 + am) * DD + ak;
  const bf16* aptr1 = aptr0 + (size_t)16 * DD;
  v8f acc0[4] = {}, acc1[4] = {};
  for (int k0 = 0; k0 < DD; k0 += 32) {
    __builtin_prefetch(aptr0 + k0 + 128, 0, 1);
    __builtin_prefetch(aptr1 + k0 + 128, 0, 1);
    v16bf a0 = cat8(*(const v8bf*)(aptr0 + k0), *(const v8bf*)(aptr0 + k0 + 16));
    v16bf a1 = cat8(*(const v8bf*)(aptr1 + k0), *(const v8bf*)(aptr1 + k0 + 16));
#pragma unroll
    for (int t = 0; t < 4; ++t) {
      const bf16* bb = w + (size_t)(n0 + t * 16 + bn) * DD + k0 + bk;
      v16bf bmat = cat8(*(const v8bf*)(bb), *(const v8bf*)(bb + 16));
      acc0[t] = __builtin_amdgcn_wmma_f32_16x16x32_bf16(
          false, a0, false, bmat, (short)0, acc0[t], false, false);
      acc1[t] = __builtin_amdgcn_wmma_f32_16x16x32_bf16(
          false, a1, false, bmat, (short)0, acc1[t], false, false);
    }
  }

  // C/D layout: lane -> col n=(lane&15); vgpr j -> row m = j + 8*(lane>>4)
  const int cn = lane & 15, cm = (lane >> 4) * 8;
  if (z == 2) {
#pragma unroll
    for (int t = 0; t < 4; ++t) {
      const int e = n0 + t * 16 + cn;
#pragma unroll
      for (int j = 0; j < 8; ++j) {
        const int mg0 = m0 + cm + j;
        const int b0 = mg0 >> 11, s0 = mg0 & (SS - 1);
        vt[((size_t)b0 * DD + e) * SS + s0] = (bf16)acc0[t][j];
        const int mg1 = mg0 + 16;
        const int b1 = mg1 >> 11, s1 = mg1 & (SS - 1);
        vt[((size_t)b1 * DD + e) * SS + s1] = (bf16)acc1[t][j];
      }
    }
  } else {
    bf16* dst = (z == 0) ? qb : kbuf;
    const float scale = (z == 0) ? 0.03125f : 1.0f;  // 1/sqrt(1024) = 1/32
#pragma unroll
    for (int t = 0; t < 4; ++t)
#pragma unroll
      for (int j = 0; j < 8; ++j) {
        dst[(size_t)(m0 + cm + j) * DD + (n0 + t * 16 + cn)] =
            (bf16)(acc0[t][j] * scale);
        dst[(size_t)(m0 + 16 + cm + j) * DD + (n0 + t * 16 + cn)] =
            (bf16)(acc1[t][j] * scale);
      }
  }
}

// ---------------- scores = Q @ K^T (causal masked) -------
__global__ __launch_bounds__(256) void scores_kernel(const bf16* __restrict__ qb,
                                                     const bf16* __restrict__ kbuf,
                                                     bf16* __restrict__ sbuf) {
  const int b = blockIdx.z;
  // Skip blocks entirely above the causal diagonal (block tile 256q x 64k).
  if ((int)blockIdx.y * 64 > (int)blockIdx.x * 256 + 255) return;

  const int lane = threadIdx.x & 31;
  const int wid  = threadIdx.x >> 5;
  const int m0   = blockIdx.x * 256 + wid * 32;   // q rows
  const int n0   = blockIdx.y * 64;               // k cols

  const int am = lane & 15, ak = (lane >> 4) * 8;
  const int bn = lane & 15, bk = (lane >> 4) * 8;

  const bf16* aptr0 = qb + ((size_t)b * SS + m0 + am) * DD + ak;
  const bf16* aptr1 = aptr0 + (size_t)16 * DD;
  v8f acc0[4] = {}, acc1[4] = {};
  for (int k0 = 0; k0 < DD; k0 += 32) {
    __builtin_prefetch(aptr0 + k0 + 128, 0, 1);
    __builtin_prefetch(aptr1 + k0 + 128, 0, 1);
    v16bf a0 = cat8(*(const v8bf*)(aptr0 + k0), *(const v8bf*)(aptr0 + k0 + 16));
    v16bf a1 = cat8(*(const v8bf*)(aptr1 + k0), *(const v8bf*)(aptr1 + k0 + 16));
#pragma unroll
    for (int t = 0; t < 4; ++t) {
      const bf16* bb = kbuf + ((size_t)b * SS + n0 + t * 16 + bn) * DD + k0 + bk;
      v16bf bmat = cat8(*(const v8bf*)(bb), *(const v8bf*)(bb + 16));
      acc0[t] = __builtin_amdgcn_wmma_f32_16x16x32_bf16(
          false, a0, false, bmat, (short)0, acc0[t], false, false);
      acc1[t] = __builtin_amdgcn_wmma_f32_16x16x32_bf16(
          false, a1, false, bmat, (short)0, acc1[t], false, false);
    }
  }

  const int cn = lane & 15, cm = (lane >> 4) * 8;
  const float NEG_INF = -__builtin_huge_valf();
#pragma unroll
  for (int t = 0; t < 4; ++t)
#pragma unroll
    for (int j = 0; j < 8; ++j) {
      const int k = n0 + t * 16 + cn;
      const int q0 = m0 + cm + j;
      const float v0 = (k <= q0) ? acc0[t][j] : NEG_INF;
      sbuf[((size_t)b * SS + q0) * SS + k] = (bf16)v0;
      const int q1 = q0 + 16;
      const float v1 = (k <= q1) ? acc1[t][j] : NEG_INF;
      sbuf[((size_t)b * SS + q1) * SS + k] = (bf16)v1;
    }
}

// ---------------- softmax rows (in place, bf16) ----------
static __device__ __forceinline__ float wred_max(float v) {
#pragma unroll
  for (int o = 16; o > 0; o >>= 1) v = fmaxf(v, __shfl_xor(v, o, 32));
  return v;
}
static __device__ __forceinline__ float wred_sum(float v) {
#pragma unroll
  for (int o = 16; o > 0; o >>= 1) v += __shfl_xor(v, o, 32);
  return v;
}

__global__ __launch_bounds__(256) void softmax_kernel(bf16* __restrict__ sbuf) {
  const int row = blockIdx.x;          // b*S + q
  const int q   = row & (SS - 1);
  bf16* p = sbuf + (size_t)row * SS;
  const int tid  = threadIdx.x;
  const int lane = tid & 31, wid = tid >> 5;
  __shared__ float red[8];

  float r[8];
  float m = -__builtin_huge_valf();
#pragma unroll
  for (int i = 0; i < 8; ++i) {
    const int k = i * 256 + tid;
    r[i] = (k <= q) ? (float)p[k] : -__builtin_huge_valf();
    m = fmaxf(m, r[i]);
  }
  m = wred_max(m);
  if (lane == 0) red[wid] = m;
  __syncthreads();
  float mx = red[0];
#pragma unroll
  for (int i = 1; i < 8; ++i) mx = fmaxf(mx, red[i]);
  __syncthreads();

  float s = 0.f;
#pragma unroll
  for (int i = 0; i < 8; ++i) {
    const float e = __expf(r[i] - mx);   // exp(-inf - mx) == 0
    r[i] = e;
    s += e;
  }
  s = wred_sum(s);
  if (lane == 0) red[wid] = s;
  __syncthreads();
  float tot = 0.f;
#pragma unroll
  for (int i = 0; i < 8; ++i) tot += red[i];
  const float inv = 1.0f / tot;

  // Zero-pad masked region up to the P@V kernel's per-wave K bound (m0+32
  // with m0 = 32-aligned query tile base; zend = align32(q)+64 covers it).
  const int zend = min(SS, (((q >> 5) + 2) << 5));
#pragma unroll
  for (int i = 0; i < 8; ++i) {
    const int k = i * 256 + tid;
    if (k <= q)            p[k] = (bf16)(r[i] * inv);
    else if (k < zend)     p[k] = (bf16)0.0f;
  }
}

// ---------------- out = P @ V  (V pre-transposed) --------
__global__ __launch_bounds__(256) void pv_kernel(const bf16* __restrict__ sbuf,
                                                 const bf16* __restrict__ vt,
                                                 float* __restrict__ out) {
  const int b    = blockIdx.z;
  const int lane = threadIdx.x & 31;
  const int wid  = threadIdx.x >> 5;
  const int m0   = blockIdx.x * 256 + wid * 32;   // q rows (32-aligned)
  const int n0   = blockIdx.y * 64;               // d cols
  const int kend = m0 + 32;                       // causal K bound

  const int am = lane & 15, ak = (lane >> 4) * 8;
  const int bn = lane & 15, bk = (lane >> 4) * 8;

  const bf16* aptr0 = sbuf + ((size_t)b * SS + m0 + am) * SS + ak;
  const bf16* aptr1 = aptr0 + (size_t)16 * SS;
  const bf16* bptr  = vt + ((size_t)b * DD + n0 + bn) * SS + bk;

  v8f acc0[4] = {}, acc1[4] = {};
  for (int k0 = 0; k0 < kend; k0 += 32) {
    __builtin_prefetch(aptr0 + k0 + 128, 0, 1);
    __builtin_prefetch(aptr1 + k0 + 128, 0, 1);
    v16bf a0 = cat8(*(const v8bf*)(aptr0 + k0), *(const v8bf*)(aptr0 + k0 + 16));
    v16bf a1 = cat8(*(const v8bf*)(aptr1 + k0), *(const v8bf*)(aptr1 + k0 + 16));
#pragma unroll
    for (int t = 0; t < 4; ++t) {
      const bf16* bb = bptr + (size_t)t * 16 * SS + k0;
      v16bf bmat = cat8(*(const v8bf*)(bb), *(const v8bf*)(bb + 16));
      acc0[t] = __builtin_amdgcn_wmma_f32_16x16x32_bf16(
          false, a0, false, bmat, (short)0, acc0[t], false, false);
      acc1[t] = __builtin_amdgcn_wmma_f32_16x16x32_bf16(
          false, a1, false, bmat, (short)0, acc1[t], false, false);
    }
  }

  const int cn = lane & 15, cm = (lane >> 4) * 8;
#pragma unroll
  for (int t = 0; t < 4; ++t)
#pragma unroll
    for (int j = 0; j < 8; ++j) {
      out[((size_t)b * SS + m0 + cm + j) * DD + (n0 + t * 16 + cn)] = acc0[t][j];
      out[((size_t)b * SS + m0 + 16 + cm + j) * DD + (n0 + t * 16 + cn)] = acc1[t][j];
    }
}

extern "C" void kernel_launch(void* const* d_in, const int* in_sizes, int n_in,
                              void* d_out, int out_size, void* d_ws, size_t ws_size,
                              hipStream_t stream) {
  const float* x  = (const float*)d_in[0];
  const float* wq = (const float*)d_in[1];
  const float* wk = (const float*)d_in[2];
  const float* wv = (const float*)d_in[3];
  float* out = (float*)d_out;

  // Workspace layout (bf16 everywhere): ~102 MB total, fits in 192 MB L2.
  char* p = (char*)d_ws;
  bf16* xbf  = (bf16*)p; p += (size_t)BB * SS * DD * sizeof(bf16);
  bf16* wqb  = (bf16*)p; p += (size_t)DD * DD * sizeof(bf16);
  bf16* wkb  = (bf16*)p; p += (size_t)DD * DD * sizeof(bf16);
  bf16* wvb  = (bf16*)p; p += (size_t)DD * DD * sizeof(bf16);
  bf16* qb   = (bf16*)p; p += (size_t)BB * SS * DD * sizeof(bf16);
  bf16* kbuf = (bf16*)p; p += (size_t)BB * SS * DD * sizeof(bf16);
  bf16* vt   = (bf16*)p; p += (size_t)BB * SS * DD * sizeof(bf16);
  bf16* sbuf = (bf16*)p; p += (size_t)BB * SS * SS * sizeof(bf16);

  const int nx = BB * SS * DD;   // 8,388,608
  const int nw = DD * DD;        // 1,048,576
  cvt_bf16_kernel<<<nx / 1024, 256, 0, stream>>>(x,  xbf, nx);
  cvt_bf16_kernel<<<nw / 1024, 256, 0, stream>>>(wq, wqb, nw);
  cvt_bf16_kernel<<<nw / 1024, 256, 0, stream>>>(wk, wkb, nw);
  cvt_bf16_kernel<<<nw / 1024, 256, 0, stream>>>(wv, wvb, nw);

  qkv_kernel<<<dim3(32, 16, 3), 256, 0, stream>>>(xbf, wqb, wkb, wvb, qb, kbuf, vt);
  scores_kernel<<<dim3(8, 32, 4), 256, 0, stream>>>(qb, kbuf, sbuf);
  softmax_kernel<<<BB * SS, 256, 0, stream>>>(sbuf);
  pv_kernel<<<dim3(8, 16, 4), 256, 0, stream>>>(sbuf, vt, out);
}
